// AttnDecoderRNN_51127290691657
// MI455X (gfx1250) — matrix operand order repaired and checked by reading, not compiled, not run on hardware
//
#include <hip/hip_runtime.h>
#include <math.h>

// ---------------------------------------------------------------------------
// Problem sizes (fixed by the reference)
// ---------------------------------------------------------------------------
constexpr int HN = 512;     // hidden
constexpr int EN = 256;     // embed
constexpr int VN = 32000;   // vocab
constexpr int LN = 1024;    // source length
constexpr int BN = 64;      // batch

typedef _Float16 half_t;
typedef __attribute__((ext_vector_type(16))) _Float16 v16h;
typedef __attribute__((ext_vector_type(8)))  float    v8f;

// Branch-free tanh on top of the native exp (v_exp_f32).  Robust at the
// extremes without clamping: exp(2x)->inf => 1, exp(2x)->0 => -1.
__device__ __forceinline__ float fast_tanhf(float x) {
    const float e = __expf(2.0f * x);
    return 1.0f - 2.0f / (e + 1.0f);
}
__device__ __forceinline__ float sigmoidf_(float x) {
    return 1.0f / (1.0f + __expf(-x));
}

// Convert 16 contiguous floats into a B-style v16h fragment
__device__ __forceinline__ v16h cvt16(const float* __restrict__ p) {
    v16h r;
#pragma unroll
    for (int i = 0; i < 16; ++i) r[i] = (half_t)p[i];
    return r;
}

// ---------------------------------------------------------------------------
// Kernel 1: embedding gather + LSTM cell.  grid = 2*B blocks of 256 threads,
// each pair of blocks covers one batch row (n in [0,512)).
// ---------------------------------------------------------------------------
__global__ void lstm_kernel(const int* __restrict__ tok,
                            const float* __restrict__ loc,   // last_output_context (B,H)
                            const float* __restrict__ hin,   // last_hidden (B,H)
                            const float* __restrict__ cin,   // last_cell (B,H)
                            const float* __restrict__ emb,   // (V,E)
                            const float* __restrict__ Wih,   // (4H, E+H)
                            const float* __restrict__ Whh,   // (4H, H)
                            const float* __restrict__ bih,
                            const float* __restrict__ bhh,
                            float* __restrict__ hout,
                            float* __restrict__ cout) {
    __shared__ float xs[EN + HN];   // x = [emb[tok], last_output_context]
    __shared__ float hs[HN];
    const int b = blockIdx.x >> 1;
    const int t = threadIdx.x;
    const int tk = tok[b];
    for (int i = t; i < EN; i += 256) xs[i] = emb[tk * EN + i];
    for (int i = t; i < HN; i += 256) {
        xs[EN + i] = loc[b * HN + i];
        hs[i] = hin[b * HN + i];
    }
    __syncthreads();
    const int n = ((blockIdx.x & 1) << 8) + t;   // 0..511
    float g4[4];
#pragma unroll
    for (int gi = 0; gi < 4; ++gi) {
        const int row = gi * HN + n;
        const float* wi = Wih + (size_t)row * (EN + HN);
        const float* wh = Whh + (size_t)row * HN;
        float s = bih[row] + bhh[row];
        for (int k = 0; k < EN + HN; k += 4) {
            float4 w = *(const float4*)(wi + k);
            s = fmaf(xs[k], w.x, fmaf(xs[k+1], w.y, fmaf(xs[k+2], w.z, fmaf(xs[k+3], w.w, s))));
        }
        for (int k = 0; k < HN; k += 4) {
            float4 w = *(const float4*)(wh + k);
            s = fmaf(hs[k], w.x, fmaf(hs[k+1], w.y, fmaf(hs[k+2], w.z, fmaf(hs[k+3], w.w, s))));
        }
        g4[gi] = s;
    }
    const float ig = sigmoidf_(g4[0]);
    const float fg = sigmoidf_(g4[1]);
    const float gg = fast_tanhf(g4[2]);
    const float og = sigmoidf_(g4[3]);
    const float c  = fg * cin[b * HN + n] + ig * gg;
    const float h  = og * fast_tanhf(c);
    hout[b * HN + n] = h;
    cout[b * HN + n] = c;
}

// ---------------------------------------------------------------------------
// Kernel 2: hW[b,n] = b_attn[n] + sum_k h[b,k] * W_attn[n,k]  (h-half of the
// attention matmul, computed once instead of per-L).  grid = 2*B x 256.
// ---------------------------------------------------------------------------
__global__ void hw_kernel(const float* __restrict__ h,
                          const float* __restrict__ Wat,   // (H, 2H)
                          const float* __restrict__ bat,
                          float* __restrict__ hW) {
    __shared__ float hs[HN];
    const int b = blockIdx.x >> 1;
    const int t = threadIdx.x;
    for (int i = t; i < HN; i += 256) hs[i] = h[b * HN + i];
    __syncthreads();
    const int n = ((blockIdx.x & 1) << 8) + t;
    const float* wr = Wat + (size_t)n * (2 * HN);   // columns [0,H): h half
    float s = bat[n];
    for (int k = 0; k < HN; k += 4) {
        float4 w = *(const float4*)(wr + k);
        s = fmaf(hs[k], w.x, fmaf(hs[k+1], w.y, fmaf(hs[k+2], w.z, fmaf(hs[k+3], w.w, s))));
    }
    hW[b * HN + n] = s;
}

// ---------------------------------------------------------------------------
// Kernel 2b: pre-convert the enc-half of W_attn (512x512 f32, reused by all
// 4096 energy waves) into f16 laid out in exact B-fragment order:
//   dst[(((nt*16)+kc)*32 + lane)*16 + j] = W_attn[n, 512 + K]
//   n = nt*16 + lane%16,  K = kc*32 + (lane/16)*16 + j
// ---------------------------------------------------------------------------
__global__ void wswz_kernel(const float* __restrict__ Wat, half_t* __restrict__ dst) {
    const int tid  = blockIdx.x * 256 + threadIdx.x;    // 0 .. 262143
    const int j    = tid & 15;
    const int lane = (tid >> 4) & 31;
    const int kc   = (tid >> 9) & 15;
    const int nt   = tid >> 13;
    const int n    = nt * 16 + (lane & 15);
    const int K    = kc * 32 + (lane >> 4) * 16 + j;
    dst[tid] = (half_t)Wat[(size_t)n * (2 * HN) + HN + K];
}

// ---------------------------------------------------------------------------
// Kernel 3: energy[m] = sum_n tanh( (enc@WattnE^T)[m,n] + hW[b,n] ) * beta[n]
// m = l*B + b, M = L*B = 65536.  One wave per 16-row tile; f16 WMMA, f32
// accumulate; 4 N-tiles per K-pass (4 independent acc chains, no spills);
// kc loop kept rolled so A stays in LDS (ds_load_b128), not scratch.
// ---------------------------------------------------------------------------
__global__ void energy_kernel(const float* __restrict__ enc,   // (L*B, H)
                              const v16h*  __restrict__ Wf,    // pre-swizzled f16 frags
                              const float* __restrict__ hW,    // (B, H)
                              const float* __restrict__ beta,  // (H)
                              float* __restrict__ energy) {    // (L*B)
    __shared__ half_t As[16 * 32 * 16];   // [kchunk][lane][16 halves] = 16KB
    const int lane = threadIdx.x;          // 0..31
    const int m0   = blockIdx.x * 16;
    const int r    = m0 + (lane & 15);     // enc row this lane stages/owns
    const int hh   = lane >> 4;            // lane half

    // Phase 1: stage A tile (16 rows x K=512) as f16 in A-fragment order.
    // A lane layout: K = kc*32 + hh*16 + j
#pragma unroll 1
    for (int kc = 0; kc < 16; ++kc) {
        const float* p0 = enc + (size_t)r * HN + kc * 32 + hh * 16;
        const float* p1 = p0 + 8;
        half_t* dst = &As[(kc * 32 + lane) * 16];
#pragma unroll
        for (int i = 0; i < 8; ++i) dst[i]     = (half_t)p0[i];
#pragma unroll
        for (int i = 0; i < 8; ++i) dst[8 + i] = (half_t)p1[i];
    }
    __syncthreads();

    float epart[8];
#pragma unroll
    for (int v = 0; v < 8; ++v) epart[v] = 0.0f;
    const int b0 = m0 & (BN - 1);          // tile never crosses an l boundary

#pragma unroll 1
    for (int np = 0; np < 8; ++np) {       // 8 passes x 4 N-tiles = 32 N-tiles
        const int ntb = np * 4;
        v8f acc[4] = {};
#pragma unroll 1
        for (int kc = 0; kc < 16; ++kc) {
            v16h a = *(const v16h*)&As[(kc * 32 + lane) * 16];
#pragma unroll
            for (int q = 0; q < 4; ++q) {
                v16h bf = Wf[((ntb + q) * 16 + kc) * 32 + lane];
                acc[q] = __builtin_amdgcn_wmma_f32_16x16x32_f16(
                    false, a, false, bf, (short)0, acc[q], false, false);
            }
        }
#pragma unroll
        for (int q = 0; q < 4; ++q) {
            const int ncol = (ntb + q) * 16 + (lane & 15);
            const float bet = beta[ncol];
#pragma unroll
            for (int v = 0; v < 8; ++v) {
                const int mrow = v + hh * 8;               // C/D: lane half holds M+8
                const float g = acc[q][v] + hW[(b0 + mrow) * HN + ncol];
                epart[v] += fast_tanhf(g) * bet;
            }
        }
    }
    // Reduce over the 16 N-columns held by each lane half.
#pragma unroll
    for (int v = 0; v < 8; ++v) {
        float e = epart[v];
        e += __shfl_xor(e, 1, 16);
        e += __shfl_xor(e, 2, 16);
        e += __shfl_xor(e, 4, 16);
        e += __shfl_xor(e, 8, 16);
        if ((lane & 15) == 0) energy[m0 + hh * 8 + v] = e;
    }
}

// ---------------------------------------------------------------------------
// Kernel 4: softmax over L per batch column (in place: energy -> attn weights)
// ---------------------------------------------------------------------------
__global__ void softmax_kernel(float* __restrict__ e) {     // (L, B)
    __shared__ float red[256];
    const int b = blockIdx.x, t = threadIdx.x;
    float m = -1e30f;
    for (int l = t; l < LN; l += 256) m = fmaxf(m, e[l * BN + b]);
    red[t] = m; __syncthreads();
    for (int s = 128; s > 0; s >>= 1) { if (t < s) red[t] = fmaxf(red[t], red[t + s]); __syncthreads(); }
    m = red[0]; __syncthreads();
    float sum = 0.0f;
    for (int l = t; l < LN; l += 256) sum += __expf(e[l * BN + b] - m);
    red[t] = sum; __syncthreads();
    for (int s = 128; s > 0; s >>= 1) { if (t < s) red[t] += red[t + s]; __syncthreads(); }
    const float inv = 1.0f / red[0];
    for (int l = t; l < LN; l += 256) e[l * BN + b] = __expf(e[l * BN + b] - m) * inv;
}

// ---------------------------------------------------------------------------
// Kernel 5: context[b,h] = sum_l attn[l,b] * enc[l,b,h]   (enc L2-hot 2nd pass)
// ---------------------------------------------------------------------------
__global__ void context_kernel(const float* __restrict__ attn,  // (L,B)
                               const float* __restrict__ enc,   // (L*B, H)
                               float* __restrict__ ctx) {       // (B, H)
    const int b = blockIdx.x;
    const int hc = threadIdx.x;                 // 512 threads
    float acc = 0.0f;
    for (int l = 0; l < LN; ++l)
        acc = fmaf(attn[l * BN + b], enc[((size_t)l * BN + b) * HN + hc], acc);
    ctx[b * HN + hc] = acc;
}

// ---------------------------------------------------------------------------
// Kernel 6: co = tanh([h, context] @ W_oc^T + b_oc)
// ---------------------------------------------------------------------------
__global__ void co_kernel(const float* __restrict__ h,
                          const float* __restrict__ ctx,
                          const float* __restrict__ Woc,   // (H, 2H)
                          const float* __restrict__ boc,
                          float* __restrict__ co) {
    __shared__ float hs[HN], cs[HN];
    const int b = blockIdx.x >> 1;
    const int t = threadIdx.x;
    for (int i = t; i < HN; i += 256) { hs[i] = h[b * HN + i]; cs[i] = ctx[b * HN + i]; }
    __syncthreads();
    const int n = ((blockIdx.x & 1) << 8) + t;
    const float* wr = Woc + (size_t)n * (2 * HN);
    float s = boc[n];
    for (int k = 0; k < HN; k += 4) {
        float4 w = *(const float4*)(wr + k);
        s = fmaf(hs[k], w.x, fmaf(hs[k+1], w.y, fmaf(hs[k+2], w.z, fmaf(hs[k+3], w.w, s))));
    }
    for (int k = 0; k < HN; k += 4) {
        float4 w = *(const float4*)(wr + HN + k);
        s = fmaf(cs[k], w.x, fmaf(cs[k+1], w.y, fmaf(cs[k+2], w.z, fmaf(cs[k+3], w.w, s))));
    }
    co[b * HN + n] = fast_tanhf(s);
}

// ---------------------------------------------------------------------------
// Kernel 7: logits = co @ W_out^T + b_out, WMMA.  M=64 (4 tiles), N=32000
// (2000 tiles), K=512.  One wave per (mtile, group of 8 ntiles); 4 N-tiles
// per K-pass, rolled kc loop (A from LDS, W_out cvt f32->f16 on the fly —
// W_out is single-use so pre-converting would add HBM traffic).
// ---------------------------------------------------------------------------
__global__ void vocab_kernel(const float* __restrict__ co,    // (B, H)
                             const float* __restrict__ Wout,  // (V, H)
                             const float* __restrict__ bout,
                             float* __restrict__ logits) {    // (B, V)
    __shared__ half_t As[16 * 32 * 16];
    const int lane = threadIdx.x;
    const int mt = blockIdx.x & 3;
    const int ng = blockIdx.x >> 2;
    const int m0 = mt * 16;
    const int r  = m0 + (lane & 15);
    const int hh = lane >> 4;

#pragma unroll 1
    for (int kc = 0; kc < 16; ++kc) {
        const float* p0 = co + (size_t)r * HN + kc * 32 + hh * 16;
        const float* p1 = p0 + 8;
        half_t* dst = &As[(kc * 32 + lane) * 16];
#pragma unroll
        for (int i = 0; i < 8; ++i) dst[i]     = (half_t)p0[i];
#pragma unroll
        for (int i = 0; i < 8; ++i) dst[8 + i] = (half_t)p1[i];
    }
    __syncthreads();

#pragma unroll 1
    for (int np = 0; np < 2; ++np) {       // 2 passes x 4 N-tiles = 8 N-tiles
        const int ntb = ng * 8 + np * 4;
        v8f acc[4] = {};
#pragma unroll 1
        for (int kc = 0; kc < 16; ++kc) {
            v16h a = *(const v16h*)&As[(kc * 32 + lane) * 16];
#pragma unroll
            for (int q = 0; q < 4; ++q) {
                const float* wp = Wout + (size_t)((ntb + q) * 16 + (lane & 15)) * HN
                                + kc * 32 + hh * 16;
                v16h bf = cvt16(wp);
                acc[q] = __builtin_amdgcn_wmma_f32_16x16x32_f16(
                    false, a, false, bf, (short)0, acc[q], false, false);
            }
        }
#pragma unroll
        for (int q = 0; q < 4; ++q) {
            const int ncol = (ntb + q) * 16 + (lane & 15);
            const float bo = bout[ncol];
#pragma unroll
            for (int v = 0; v < 8; ++v) {
                const int row = m0 + v + hh * 8;
                logits[(size_t)row * VN + ncol] = acc[q][v] + bo;
            }
        }
    }
}

// ---------------------------------------------------------------------------
// Kernel 8: in-place log_softmax over V per batch row.
// ---------------------------------------------------------------------------
__global__ void lsm_kernel(float* __restrict__ out) {   // (B, V)
    __shared__ float red[256];
    const int b = blockIdx.x, t = threadIdx.x;
    float* row = out + (size_t)b * VN;
    float m = -1e30f;
    for (int v = t; v < VN; v += 256) m = fmaxf(m, row[v]);
    red[t] = m; __syncthreads();
    for (int s = 128; s > 0; s >>= 1) { if (t < s) red[t] = fmaxf(red[t], red[t + s]); __syncthreads(); }
    m = red[0]; __syncthreads();
    float sum = 0.0f;
    for (int v = t; v < VN; v += 256) sum += __expf(row[v] - m);
    red[t] = sum; __syncthreads();
    for (int s = 128; s > 0; s >>= 1) { if (t < s) red[t] += red[t + s]; __syncthreads(); }
    const float lse = m + logf(red[0]);
    for (int v = t; v < VN; v += 256) row[v] = row[v] - lse;
}

// ---------------------------------------------------------------------------
extern "C" void kernel_launch(void* const* d_in, const int* in_sizes, int n_in,
                              void* d_out, int out_size, void* d_ws, size_t ws_size,
                              hipStream_t stream) {
    (void)in_sizes; (void)n_in; (void)out_size; (void)ws_size;
    const int*   tok  = (const int*)  d_in[0];
    const float* loc  = (const float*)d_in[1];
    const float* hin  = (const float*)d_in[2];
    const float* cin  = (const float*)d_in[3];
    const float* enc  = (const float*)d_in[4];
    const float* emb  = (const float*)d_in[5];
    const float* Wih  = (const float*)d_in[6];
    const float* Whh  = (const float*)d_in[7];
    const float* bih  = (const float*)d_in[8];
    const float* bhh  = (const float*)d_in[9];
    const float* Wat  = (const float*)d_in[10];
    const float* bat  = (const float*)d_in[11];
    const float* beta = (const float*)d_in[12];
    const float* Woc  = (const float*)d_in[13];
    const float* boc  = (const float*)d_in[14];
    const float* Wout = (const float*)d_in[15];
    const float* bout = (const float*)d_in[16];

    // Output tuple, concatenated flat: output (B,V), co (1,B,H), h (1,B,H), c (1,B,H)
    float* out0 = (float*)d_out;
    float* oco  = out0 + (size_t)BN * VN;
    float* oh   = oco + BN * HN;
    float* ocst = oh  + BN * HN;

    // Workspace: hW (B*H) | energy/attn (L*B) | context (B*H) | Wf16 (512KB)
    float*  wsf   = (float*)d_ws;
    float*  hWbuf = wsf;
    float*  ebuf  = wsf + BN * HN;
    float*  ctx   = ebuf + LN * BN;
    half_t* Wf16  = (half_t*)(ctx + BN * HN);

    lstm_kernel   <<<BN * 2,          256, 0, stream>>>(tok, loc, hin, cin, emb, Wih, Whh, bih, bhh, oh, ocst);
    hw_kernel     <<<BN * 2,          256, 0, stream>>>(oh, Wat, bat, hWbuf);
    wswz_kernel   <<<(HN * HN) / 256, 256, 0, stream>>>(Wat, Wf16);
    energy_kernel <<<(LN * BN) / 16,   32, 0, stream>>>(enc, (const v16h*)Wf16, hWbuf, beta, ebuf);
    softmax_kernel<<<BN,              256, 0, stream>>>(ebuf);
    context_kernel<<<BN,               HN, 0, stream>>>(ebuf, enc, ctx);
    co_kernel     <<<BN * 2,          256, 0, stream>>>(oh, ctx, Woc, boc, oco);
    vocab_kernel  <<<(VN / 128) * 4,   32, 0, stream>>>(oco, Wout, bout, out0);
    lsm_kernel    <<<BN,              256, 0, stream>>>(out0);
}
